// RENBase_52432960749673
// MI455X (gfx1250) — matrix operand order repaired and checked by reading, not compiled, not run on hardware
//
#include <hip/hip_runtime.h>

typedef _Float16 v16h __attribute__((ext_vector_type(16)));
typedef _Float16 v8h  __attribute__((ext_vector_type(8)));
typedef float    v8f  __attribute__((ext_vector_type(8)));
typedef float    vf4  __attribute__((ext_vector_type(4)));
typedef unsigned int u32x4 __attribute__((ext_vector_type(4)));
typedef int      i32x4 __attribute__((ext_vector_type(4)));
typedef int      i32x8 __attribute__((ext_vector_type(8)));

#define T_STEPS 512
#define BATCH   512
#define NU 64
#define NX 128
#define NV 256
#define NY 64

// f16 weight offsets inside d_ws (element counts)
#define OFF_A    0
#define OFF_B1   16384
#define OFF_B2   49152
#define OFF_C1   57344
#define OFF_C2   90112
#define OFF_D11  98304
#define OFF_D12  163840
#define OFF_D21  180224
#define OFF_D22  196608
#define W_TOTAL  200704

// ---------------- prep: convert f32 weights -> f16 packed in ws ----------------
__global__ void ren_prep(const float* A, const float* B1, const float* B2,
                         const float* C1, const float* C2, const float* D11,
                         const float* D12, const float* D21, const float* D22,
                         _Float16* w) {
  int i = blockIdx.x * 256 + threadIdx.x;
  if (i >= W_TOTAL) return;
  const float* s; int o;
  if      (i < OFF_B1)  { s = A;   o = i; }
  else if (i < OFF_B2)  { s = B1;  o = i - OFF_B1; }
  else if (i < OFF_C1)  { s = B2;  o = i - OFF_B2; }
  else if (i < OFF_C2)  { s = C1;  o = i - OFF_C1; }
  else if (i < OFF_D11) { s = C2;  o = i - OFF_C2; }
  else if (i < OFF_D12) { s = D11; o = i - OFF_D11; }
  else if (i < OFF_D21) { s = D12; o = i - OFF_D12; }
  else if (i < OFF_D22) { s = D21; o = i - OFF_D21; }
  else                  { s = D22; o = i - OFF_D22; }
  w[i] = (_Float16)s[o];
}

// ---------------- helpers ----------------
union H16 { v16h v; v8h h[2]; };

__device__ __forceinline__ void wave_sync() {
  __builtin_amdgcn_wave_barrier();
  __asm__ __volatile__("" ::: "memory");
}

__device__ __forceinline__ v8f wmma_f16(v16h a, v16h b, v8f c) {
  return __builtin_amdgcn_wmma_f32_16x16x32_f16(false, a, false, b, (short)0, c,
                                                false, false);
}

__device__ __forceinline__ v8f splat8(float x) {
  v8f c;
#pragma unroll
  for (int e = 0; e < 8; ++e) c[e] = x;
  return c;
}

__device__ __forceinline__ v16h zero16() {
  v16h z;
#pragma unroll
  for (int e = 0; e < 16; ++e) z[e] = (_Float16)0.0f;
  return z;
}

// B fragment (K=32 x N=16) of Bmat[k,n] = W[rbase+n][kc*32+k], W row-major [out][in].
__device__ __forceinline__ v16h ldB(const _Float16* W, int ldk, int rbase, int kc,
                                    int lane) {
  const _Float16* p =
      W + (size_t)(rbase + (lane & 15)) * ldk + kc * 32 + ((lane >> 4) << 3);
  H16 u;
  u.h[0] = *(const v8h*)p;
  u.h[1] = *(const v8h*)(p + 16);
  return u.v;
}

// A fragment (M=16 x K=32) from an f16 row-major [16][ldk] staging buffer.
__device__ __forceinline__ v16h ldA_h(const _Float16* X, int ldk, int kc, int lane) {
  int b0 = (lane >> 4) << 3;
  const _Float16* p = X + (lane & 15) * ldk + kc * 32 + b0;
  H16 u;
  u.h[0] = *(const v8h*)p;
  u.h[1] = *(const v8h*)(p + 16);
  return u.v;
}

// A fragment assembled from f32 global rows (used for u), row stride NU.
__device__ __forceinline__ v16h ldA_f32(const float* X, int kc, int lane) {
  int b0 = (lane >> 4) << 3;
  const float* p = X + (lane & 15) * NU + kc * 32 + b0;
  vf4 q0 = *(const vf4*)p;
  vf4 q1 = *(const vf4*)(p + 4);
  vf4 q2 = *(const vf4*)(p + 16);
  vf4 q3 = *(const vf4*)(p + 20);
  H16 u;
#pragma unroll
  for (int e = 0; e < 4; ++e) {
    u.h[0][e]     = (_Float16)q0[e];
    u.h[0][e + 4] = (_Float16)q1[e];
    u.h[1][e]     = (_Float16)q2[e];
    u.h[1][e + 4] = (_Float16)q3[e];
  }
  return u.v;
}

// TDM: DMA the 256x256 f16 D11 panel (global, row-major) into LDS offset 0.
// D# built per CDNA5 ISA 8.3/8.4: count=1, data_size=2B, tensor=tile=256x256,
// dim0 stride 256.  groups 2/3 zero (2D tensor).
__device__ __forceinline__ void tdm_load_d11(const _Float16* gsrc) {
  unsigned long long ga = (unsigned long long)gsrc;
  u32x4 g0;
  g0[0] = 1u;                                    // count=1, user mode
  g0[1] = 0u;                                    // lds_addr = 0 (d11s at base)
  g0[2] = (unsigned)(ga & 0xFFFFFFFFu);          // global_addr[31:0]
  g0[3] = (unsigned)((ga >> 32) & 0x01FFFFFFu)   // global_addr[56:32]
          | (2u << 30);                          // type = 2 ("image")
  i32x8 g1;
  g1[0] = (int)(1u << 16);          // workgroup_mask=0, data_size=1 (2 bytes)
  g1[1] = (int)(256u << 16);        // tensor_dim0[15:0] in bits[63:48]
  g1[2] = (int)(256u << 16);        // tensor_dim0 hi=0 | tensor_dim1[15:0]
  g1[3] = (int)(256u << 16);        // tensor_dim1 hi=0 | tile_dim0=256
  g1[4] = 256;                      // tile_dim1=256, tile_dim2=0
  g1[5] = 256;                      // tensor_dim0_stride lo32
  g1[6] = 0;                        // dim0_stride hi | dim1_stride lo
  g1[7] = 0;
  i32x4 z4 = {0, 0, 0, 0};
#if __clang_major__ >= 23
  i32x8 z8 = {0, 0, 0, 0, 0, 0, 0, 0};
  __builtin_amdgcn_tensor_load_to_lds(g0, g1, z4, z4, z8, 0);
#else
  __builtin_amdgcn_tensor_load_to_lds(g0, g1, z4, z4, 0);
#endif
}

// ---------------- LDS layout (bytes) ----------------
#define L_D11   0        // 65536 f16  -> 131072 (TDM target; must stay at 0)
#define L_C1    131072   // 32768 f16  ->  65536
#define L_D12   196608   // 16384 f16  ->  32768
#define L_D21   229376   // 16384 f16  ->  32768
#define L_DIAG  262144   // 4096 f32   ->  16384  (16 diag blocks of D11, f32)
#define L_BV    278528   // 256 f32
#define L_BX    279552   // 128 f32
#define L_BY    280064   // 64 f32
#define L_WAVE  280320   // 8 waves x 4096 B (xstage; solve/solveh aliased inside)
#define L_TOTAL (280320 + 8 * 4096)   // 313,088 B

// ---------------- main rollout kernel ----------------
// grid = 4 workgroups x 256 threads (8 waves); each wave owns 16 batch rows.
__global__ __launch_bounds__(256) void ren_rollout(
    const float* __restrict__ x0, const float* __restrict__ u,
    const float* __restrict__ D11f, const float* __restrict__ bx,
    const float* __restrict__ bv, const float* __restrict__ by,
    const _Float16* __restrict__ w, float* __restrict__ xout,
    float* __restrict__ yout) {
  extern __shared__ char smem[];
  _Float16* d11s = (_Float16*)(smem + L_D11);
  _Float16* c1s  = (_Float16*)(smem + L_C1);
  _Float16* d12s = (_Float16*)(smem + L_D12);
  _Float16* d21s = (_Float16*)(smem + L_D21);
  float* diag    = (float*)(smem + L_DIAG);
  float* bvs     = (float*)(smem + L_BV);
  float* bxs     = (float*)(smem + L_BX);
  float* bys     = (float*)(smem + L_BY);
  char* wbase    = smem + L_WAVE + (threadIdx.x >> 5) * 4096;
  _Float16* xstage = (_Float16*)wbase;            // 16x128 f16 (4096 B)
  float* solve     = (float*)wbase;               // 16x16 f32 (aliased; idle phases)
  _Float16* solveh = (_Float16*)(wbase + 1024);   // 16x16 f16 (aliased)

  const int tid = threadIdx.x;
  const int lane = tid & 31;
  const int nlane = lane & 15;
  const int mrow = (lane >> 4) << 3;  // C-layout row offset for this half-wave
  const int r0 = blockIdx.x * 128 + (tid >> 5) * 16;

  const _Float16* Ah  = w + OFF_A;
  const _Float16* B1h = w + OFF_B1;
  const _Float16* B2h = w + OFF_B2;
  const _Float16* C2h = w + OFF_C2;
  const _Float16* D22h = w + OFF_D22;

  // ---- preload LDS: D11 via Tensor Data Mover, rest via copy loops ----
  if (tid < 32) tdm_load_d11(w + OFF_D11);
  for (int i = tid; i < NV * NX; i += 256) c1s[i] = w[OFF_C1 + i];
  for (int i = tid; i < NV * NU; i += 256) d12s[i] = w[OFF_D12 + i];
  for (int i = tid; i < NY * NV; i += 256) d21s[i] = w[OFF_D21 + i];
  for (int i = tid; i < 16 * 256; i += 256) {
    int blk = i >> 8, ii = (i >> 4) & 15, jj = i & 15;
    diag[i] = D11f[(size_t)(blk * 16 + ii) * NV + blk * 16 + jj];
  }
  for (int i = tid; i < NV; i += 256) bvs[i] = bv[i];
  if (tid < NX) bxs[tid] = bx[tid];
  if (tid < NY) bys[tid] = by[tid];
  if (tid < 32) __builtin_amdgcn_s_wait_tensorcnt((short)0);
  __syncthreads();

  // x0 -> per-wave f16 staging, then into A fragments
  for (int i = lane; i < 16 * NX; i += 32)
    xstage[i] = (_Float16)x0[(size_t)(r0 + (i >> 7)) * NX + (i & 127)];
  wave_sync();

  v16h xf[4], wf[8], uf[2];
#pragma unroll
  for (int kc = 0; kc < 4; ++kc) xf[kc] = ldA_h(xstage, NX, kc, lane);
  wave_sync();

  const v16h zf = zero16();

  for (int t = 0; t < T_STEPS; ++t) {
#pragma unroll
    for (int f = 0; f < 8; ++f) wf[f] = zf;  // w0 = 0 each step

    const float* ubase = u + ((size_t)t * BATCH + r0) * NU;
    uf[0] = ldA_f32(ubase, 0, lane);
    uf[1] = ldA_f32(ubase, 1, lane);
    if (t + 1 < T_STEPS) {  // prefetch next step's u tile (4 KB, 32 x 128 B)
      const char* pn =
          (const char*)(u + ((size_t)(t + 1) * BATCH + r0) * NU) + lane * 128;
      __builtin_prefetch(pn, 0, 1);
    }

    // ---- equilibrium, software-pipelined over 16 column blocks ----
    v8f cnext = splat8(bvs[nlane]);
#pragma unroll
    for (int kc = 0; kc < 4; ++kc)
      cnext = wmma_f16(xf[kc], ldB(c1s, NX, 0, kc, lane), cnext);
#pragma unroll
    for (int kc = 0; kc < 2; ++kc)
      cnext = wmma_f16(uf[kc], ldB(d12s, NU, 0, kc, lane), cnext);

#pragma unroll
    for (int jb = 0; jb < 16; ++jb) {
      v8f c = cnext;
      if (jb >= 1) {
        const int lf = (jb - 1) >> 1;  // "late" fragment needing block jb-1
        c = wmma_f16(wf[lf], ldB(d11s, NV, jb * 16, lf, lane), c);
      }
#pragma unroll
      for (int v = 0; v < 8; ++v) solve[(v + mrow) * 16 + nlane] = c[v];
      wave_sync();

      // block jb+1's independent WMMAs co-execute with the serial solve
      if (jb < 15) {
        const int m = jb + 1;
        cnext = splat8(bvs[m * 16 + nlane]);
#pragma unroll
        for (int kc = 0; kc < 4; ++kc)
          cnext = wmma_f16(xf[kc], ldB(c1s, NX, m * 16, kc, lane), cnext);
#pragma unroll
        for (int kc = 0; kc < 2; ++kc)
          cnext = wmma_f16(uf[kc], ldB(d12s, NU, m * 16, kc, lane), cnext);
        const int ne = jb >> 1;  // fully-solved w fragments
        for (int fj = 0; fj < ne; ++fj)
          cnext = wmma_f16(wf[fj], ldB(d11s, NV, m * 16, fj, lane), cnext);
      }

      // serial 16-step in-block solve; lane r handles batch row r
      if (lane < 16) {
        float a[16], ws_[16];
#pragma unroll
        for (int q = 0; q < 4; ++q) {
          vf4 tq = *(const vf4*)(solve + lane * 16 + q * 4);
          a[q * 4 + 0] = tq[0]; a[q * 4 + 1] = tq[1];
          a[q * 4 + 2] = tq[2]; a[q * 4 + 3] = tq[3];
        }
        const float* dblk = diag + jb * 256;
        H16 hv;
#pragma unroll
        for (int i = 0; i < 16; ++i) {
          float acc0 = a[i], acc1 = 0.0f;
          int j = 0;
          for (; j + 1 < i; j += 2) {
            acc0 = fmaf(dblk[i * 16 + j], ws_[j], acc0);
            acc1 = fmaf(dblk[i * 16 + j + 1], ws_[j + 1], acc1);
          }
          if (j < i) acc0 = fmaf(dblk[i * 16 + j], ws_[j], acc0);
          ws_[i] = fmaxf(acc0 + acc1, 0.0f);
          hv.v[i] = (_Float16)ws_[i];
        }
        *(v16h*)(solveh + lane * 16) = hv.v;  // packed 32-B store
      }
      wave_sync();

      // pack solved block into the w A-fragment (half selected by jb&1)
      v8h part = *(const v8h*)(solveh + nlane * 16 + mrow);
      H16 W;
      W.v = wf[jb >> 1];
      W.h[jb & 1] = part;
      wf[jb >> 1] = W.v;
    }

    // ---- x1 = x@A^T + w@B1^T + u@B2^T + bx : pipelined tile pairs ----
#pragma unroll
    for (int np = 0; np < 4; ++np) {
      const int n0 = np * 32, n1 = n0 + 16;
      v8f c0 = splat8(bxs[n0 + nlane]);
      v8f c1 = splat8(bxs[n1 + nlane]);
      v16h b0 = ldB(Ah, NX, n0, 0, lane);
      v16h b1 = ldB(Ah, NX, n1, 0, lane);
#pragma unroll
      for (int s = 0; s < 14; ++s) {
        v16h a = (s < 4) ? xf[s] : (s < 12) ? wf[s - 4] : uf[s - 12];
        v16h nb0 = b0, nb1 = b1;
        if (s + 1 < 14) {  // issue next step's B loads before this step's WMMAs
          if (s + 1 < 4) {
            nb0 = ldB(Ah, NX, n0, s + 1, lane);
            nb1 = ldB(Ah, NX, n1, s + 1, lane);
          } else if (s + 1 < 12) {
            nb0 = ldB(B1h, NV, n0, s - 3, lane);
            nb1 = ldB(B1h, NV, n1, s - 3, lane);
          } else {
            nb0 = ldB(B2h, NU, n0, s - 11, lane);
            nb1 = ldB(B2h, NU, n1, s - 11, lane);
          }
        }
        c0 = wmma_f16(a, b0, c0);
        c1 = wmma_f16(a, b1, c1);
        b0 = nb0; b1 = nb1;
      }
      if (t == T_STEPS - 1) {
#pragma unroll
        for (int v = 0; v < 8; ++v) {
          xout[(size_t)(r0 + v + mrow) * NX + n0 + nlane] = c0[v];
          xout[(size_t)(r0 + v + mrow) * NX + n1 + nlane] = c1[v];
        }
      }
#pragma unroll
      for (int v = 0; v < 8; ++v) {
        xstage[(v + mrow) * NX + n0 + nlane] = (_Float16)c0[v];
        xstage[(v + mrow) * NX + n1 + nlane] = (_Float16)c1[v];
      }
    }

    // ---- y = x@C2^T + w@D21^T + u@D22^T + by : pipelined tile pairs ----
    float* yrow = yout + ((size_t)t * BATCH + r0) * NY;
#pragma unroll
    for (int np = 0; np < 2; ++np) {
      const int n0 = np * 32, n1 = n0 + 16;
      v8f c0 = splat8(bys[n0 + nlane]);
      v8f c1 = splat8(bys[n1 + nlane]);
      v16h b0 = ldB(C2h, NX, n0, 0, lane);
      v16h b1 = ldB(C2h, NX, n1, 0, lane);
#pragma unroll
      for (int s = 0; s < 14; ++s) {
        v16h a = (s < 4) ? xf[s] : (s < 12) ? wf[s - 4] : uf[s - 12];
        v16h nb0 = b0, nb1 = b1;
        if (s + 1 < 14) {
          if (s + 1 < 4) {
            nb0 = ldB(C2h, NX, n0, s + 1, lane);
            nb1 = ldB(C2h, NX, n1, s + 1, lane);
          } else if (s + 1 < 12) {
            nb0 = ldB(d21s, NV, n0, s - 3, lane);
            nb1 = ldB(d21s, NV, n1, s - 3, lane);
          } else {
            nb0 = ldB(D22h, NU, n0, s - 11, lane);
            nb1 = ldB(D22h, NU, n1, s - 11, lane);
          }
        }
        c0 = wmma_f16(a, b0, c0);
        c1 = wmma_f16(a, b1, c1);
        b0 = nb0; b1 = nb1;
      }
#pragma unroll
      for (int v = 0; v < 8; ++v) {
        yrow[(size_t)(v + mrow) * NY + n0 + nlane] = c0[v];
        yrow[(size_t)(v + mrow) * NY + n1 + nlane] = c1[v];
      }
    }

    // refresh x fragments from staged x1 (f16)
    wave_sync();
#pragma unroll
    for (int kc = 0; kc < 4; ++kc) xf[kc] = ldA_h(xstage, NX, kc, lane);
    wave_sync();
  }
}

extern "C" void kernel_launch(void* const* d_in, const int* in_sizes, int n_in,
                              void* d_out, int out_size, void* d_ws,
                              size_t ws_size, hipStream_t stream) {
  (void)in_sizes; (void)n_in; (void)out_size; (void)ws_size;
  const float* x0  = (const float*)d_in[0];
  const float* u   = (const float*)d_in[1];
  const float* A   = (const float*)d_in[2];
  const float* B1  = (const float*)d_in[3];
  const float* B2  = (const float*)d_in[4];
  const float* C1  = (const float*)d_in[5];
  const float* C2  = (const float*)d_in[6];
  const float* D11 = (const float*)d_in[7];
  const float* D12 = (const float*)d_in[8];
  const float* D21 = (const float*)d_in[9];
  const float* D22 = (const float*)d_in[10];
  const float* bx  = (const float*)d_in[11];
  const float* bv  = (const float*)d_in[12];
  const float* by  = (const float*)d_in[13];

  _Float16* wh = (_Float16*)d_ws;
  float* out = (float*)d_out;

  ren_prep<<<(W_TOTAL + 255) / 256, 256, 0, stream>>>(A, B1, B2, C1, C2, D11,
                                                      D12, D21, D22, wh);

  ren_rollout<<<4, 256, L_TOTAL, stream>>>(x0, u, D11, bx, bv, by, wh, out,
                                           out + (size_t)BATCH * NX);
}
